// EdgeDegreeEmbedding_55516747268849
// MI455X (gfx1250) — matrix operand order repaired and checked by reading, not compiled, not run on hardware
//
#include <hip/hip_runtime.h>
#include <hip/hip_bf16.h>

// ---------------------------------------------------------------------------
// EdgeDegreeEmbedding for MI455X (gfx1250, wave32, WMMA)
//
// out[n, i, c] = (1/RESCALE) * sum_{e: tgt(e)=n} sum_{l=0..6}
//                wigner_inv[e, i, l*(l+1)] * mlp(e)[l, c]
// mlp(e) = (silu(LN(silu(LN(x W1 + b1)) W2 + b2)) W3 + b3).reshape(7, 32)
// x(e)   = [edge_distance(128) | src_emb(64) | tgt_emb(64)]
//
// Roofline: wigner stream (768 MB) bounds the kernel (~33us @ 23.3 TB/s).
// MLP (5.6 GFLOP) runs on v_wmma_f32_16x16x32_f16 with operands pre-swizzled
// into fragment order (contiguous 32B per lane -> b128 loads, no half-merges).
// The wigner gather is issued as GLOBAL_LOAD_ASYNC_TO_LDS_B32 *before* the
// MLP so the HBM stream overlaps the WMMA work; einsum (1.76 GFLOP, K=7)
// stays on f32 VALU; scatter via global_atomic_add_f32.
// ---------------------------------------------------------------------------

typedef __attribute__((ext_vector_type(16))) _Float16 v16h;
typedef __attribute__((ext_vector_type(8)))  float    v8f;

#define EDGES_PER_BLOCK 16
#define BLOCK_THREADS   128   // 4 wave32
#define D_DIST   128
#define H_DIM    64
#define D_IN     256          // 128 + 64 + 64
#define M0_ROWS  7
#define C_SPH    32
#define D_OUT3   224          // 7*32
#define NCOEF    49
#define WCOLS    (NCOEF * M0_ROWS)   // 343 gathered wigner values per edge
#define OUT_PER_NODE 1568     // 49*32
#define RESCALE_F 5.862f

#if __has_builtin(__builtin_amdgcn_global_load_async_to_lds_b32) && \
    __has_builtin(__builtin_amdgcn_s_wait_asynccnt)
#define USE_ASYNC_LDS 1
typedef __attribute__((address_space(1))) int gint_t;
typedef __attribute__((address_space(3))) int lint_t;
#else
#define USE_ASYNC_LDS 0
#endif

// ---- WMMA fragment layout (CDNA5 ISA 7.12.2, 16-bit operands) --------------
// A (16x32 f16): slot(lane, h) holds element (m = lane&15,
//   k = (h<8?0:16) + (lane>>4)*8 + (h&7))        [h bit3 -> k bit4]
// B (32x16 f16): slot(lane, h) holds element (k = (h<8?0:16) + (lane>>4)*8 + (h&7),
//   n = lane&15)
// C/D (16x16 f32): lane&15 = n, VGPR r -> m = r + 8*(lane>>4)
//
// Fragments are stored contiguously: frag[tile][lane][h] so each lane loads
// its 32 bytes with two b128 loads.

// inverse A map: element (m, k within 32-block) -> (lane, h)
__device__ __forceinline__ void store_A_elem(_Float16* fragBase, int m, int k,
                                             _Float16 v) {
  const int kt  = k >> 5, kin = k & 31;
  const int hi  = (kin >> 3) & 1;
  const int h   = ((kin >> 4) << 3) | (kin & 7);
  const int lane = m + (hi << 4);
  fragBase[(kt * 32 + lane) * 16 + h] = v;
}

// inverse B map used by the weight-staging kernel
__device__ __forceinline__ void stage_B_elem(_Float16* dst, int K, int N,
                                             int i, float val) {
  const int k = i / N, n = i - k * N;
  const int kt = k >> 5, kin = k & 31;
  const int hi = (kin >> 3) & 1, h = ((kin >> 4) << 3) | (kin & 7);
  const int ntile = n >> 4, nin = n & 15;
  const int lane = nin + (hi << 4);
  dst[((ntile * (K >> 5) + kt) * 32 + lane) * 16 + h] = (_Float16)val;
}

// One 16x16 tile of C = A(16 x 32*KT) * B(32*KT x 16*ntiles), fragment inputs.
template <int KT>
__device__ __forceinline__ void wmma_tile(const v16h* Af /* LDS */,
                                          const v16h* Bf /* global */,
                                          int ntile, float* C, int ldc,
                                          int lane, const float* bias) {
  v8f c = {};
#pragma unroll
  for (int kt = 0; kt < KT; ++kt) {
    v16h a = Af[kt * 32 + lane];
    v16h b = Bf[(ntile * KT + kt) * 32 + lane];
    c = __builtin_amdgcn_wmma_f32_16x16x32_f16(false, a, false, b, (short)0, c,
                                               false, false);
  }
  const int n0 = ntile * 16;
  const int n = lane & 15, hi = lane >> 4;
#pragma unroll
  for (int r = 0; r < 8; ++r) {
    float v = c[r];
    if (bias) v += bias[n0 + n];
    C[(r + 8 * hi) * ldc + n0 + n] = v;
  }
}

// ---- weight staging: f32 row-major -> f16 fragment order -------------------
// ws layout (halves): [W1f: 16384][W2f: 4096][W3f: 14336] = 34816 halves
#define W1_ELEMS 16384
#define W2_ELEMS 4096
#define W3_ELEMS 14336
#define WS_HALVES (W1_ELEMS + W2_ELEMS + W3_ELEMS)

__global__ void prep_weights_f16(const float* __restrict__ W1,
                                 const float* __restrict__ W2,
                                 const float* __restrict__ W3,
                                 _Float16* __restrict__ ws) {
  int i = blockIdx.x * blockDim.x + threadIdx.x;
  if (i < W1_ELEMS) {
    stage_B_elem(ws, D_IN, H_DIM, i, W1[i]);
  } else if (i < W1_ELEMS + W2_ELEMS) {
    stage_B_elem(ws + W1_ELEMS, H_DIM, H_DIM, i - W1_ELEMS, W2[i - W1_ELEMS]);
  } else if (i < WS_HALVES) {
    const int j = i - (W1_ELEMS + W2_ELEMS);
    stage_B_elem(ws + W1_ELEMS + W2_ELEMS, H_DIM, D_OUT3, j, W3[j]);
  }
}

// ---- fused MLP + einsum + scatter ------------------------------------------
__global__ void __launch_bounds__(BLOCK_THREADS)
edge_degree_fused(const int*   __restrict__ atomic_numbers,
                  const int*   __restrict__ edge_index,   // (2, E)
                  const float* __restrict__ edge_distance,// (E, 128)
                  const float* __restrict__ wigner,       // (E, 49, 49)
                  const float* __restrict__ src_emb,      // (90, 64)
                  const float* __restrict__ tgt_emb,      // (90, 64)
                  const float* __restrict__ b1,
                  const float* __restrict__ g1,
                  const float* __restrict__ be1,
                  const float* __restrict__ b2,
                  const float* __restrict__ g2,
                  const float* __restrict__ be2,
                  const float* __restrict__ b3,
                  const _Float16* __restrict__ ws,        // staged f16 weights
                  float* __restrict__ out,                // (N_NODES, 49, 32)
                  int E) {
  // A operands in fragment order: [kt][lane][h]
  __shared__ v16h  xAf[(D_IN / 32) * 32];        // 8 KB : GEMM1 A fragments
  __shared__ v16h  x2f[(H_DIM / 32) * 32];       // 2 KB : GEMM2/3 A fragments
  __shared__ float hbuf[EDGES_PER_BLOCK][H_DIM]; // 4 KB : GEMM1/2 output
  __shared__ float mlp_s[EDGES_PER_BLOCK][D_OUT3]; // 14 KB : GEMM3 output
  __shared__ float wall[EDGES_PER_BLOCK][NCOEF * 8]; // 24.5 KB : wigner cols
  __shared__ int   tnode[EDGES_PER_BLOCK];

  const v16h* W1f = (const v16h*)ws;
  const v16h* W2f = (const v16h*)(ws + W1_ELEMS);
  const v16h* W3f = (const v16h*)(ws + W1_ELEMS + W2_ELEMS);

  const int tid  = threadIdx.x;
  const int wave = tid >> 5;
  const int lane = tid & 31;
  const int e0   = blockIdx.x * EDGES_PER_BLOCK;

  // -- 1) gather [dist | src | tgt] -> f16 fragments in LDS -----------------
  if (tid < EDGES_PER_BLOCK)
    tnode[tid] = (e0 + tid < E) ? edge_index[E + e0 + tid] : -1;

  for (int idx = tid; idx < EDGES_PER_BLOCK * D_IN; idx += BLOCK_THREADS) {
    const int m = idx >> 8, col = idx & 255;
    const int eg = e0 + m;
    float v = 0.f;
    if (eg < E) {
      if (col < D_DIST) {
        v = edge_distance[(size_t)eg * D_DIST + col];
      } else if (col < D_DIST + H_DIM) {
        const int s = edge_index[eg];            // row 0
        v = src_emb[atomic_numbers[s] * H_DIM + (col - D_DIST)];
      } else {
        const int t = edge_index[E + eg];        // row 1
        v = tgt_emb[atomic_numbers[t] * H_DIM + (col - D_DIST - H_DIM)];
      }
    }
    store_A_elem((_Float16*)xAf, m, col, (_Float16)v);
  }

  // -- 2) issue the wigner column gather for ALL 16 edges NOW, async to LDS,
  //       so the dominant HBM stream overlaps the WMMA MLP below. -----------
  for (int idx = tid; idx < EDGES_PER_BLOCK * WCOLS; idx += BLOCK_THREADS) {
    const int el = idx / WCOLS, r = idx - el * WCOLS;
    const int i = r / M0_ROWS, l = r - i * M0_ROWS;
    const int eg = e0 + el;
    if (eg < E) {
      const float* g = &wigner[(size_t)eg * (NCOEF * NCOEF) + i * NCOEF +
                               l * (l + 1)];
#if USE_ASYNC_LDS
      __builtin_amdgcn_global_load_async_to_lds_b32(
          (gint_t*)g, (lint_t*)&wall[el][i * 8 + l], 0, 0);
#else
      wall[el][i * 8 + l] = *g;
#endif
    }
  }
  __syncthreads();   // xAf visible to all waves

  // -- 3) GEMM1: (16x256) x (256x64), one 16x16 N-tile per wave -------------
  wmma_tile<8>(xAf, W1f, wave, &hbuf[0][0], H_DIM, lane, nullptr);
  __syncthreads();

  // -- 4) LN1 (+b1) + SiLU -> x2 fragments ----------------------------------
  if (tid < EDGES_PER_BLOCK) {
    float s = 0.f, s2 = 0.f;
    for (int j = 0; j < H_DIM; ++j) {
      const float v = hbuf[tid][j] + b1[j];
      s += v; s2 += v * v;
    }
    const float mu = s * (1.f / H_DIM);
    const float var = s2 * (1.f / H_DIM) - mu * mu;
    const float rs = rsqrtf(var + 1e-5f);
    for (int j = 0; j < H_DIM; ++j) {
      const float v = hbuf[tid][j] + b1[j];
      const float y = (v - mu) * rs * g1[j] + be1[j];
      store_A_elem((_Float16*)x2f, tid, j, (_Float16)(y / (1.f + expf(-y))));
    }
  }
  __syncthreads();

  // -- 5) GEMM2: (16x64) x (64x64) ------------------------------------------
  wmma_tile<2>(x2f, W2f, wave, &hbuf[0][0], H_DIM, lane, nullptr);
  __syncthreads();

  // -- 6) LN2 (+b2) + SiLU -> x2 fragments ----------------------------------
  if (tid < EDGES_PER_BLOCK) {
    float s = 0.f, s2 = 0.f;
    for (int j = 0; j < H_DIM; ++j) {
      const float v = hbuf[tid][j] + b2[j];
      s += v; s2 += v * v;
    }
    const float mu = s * (1.f / H_DIM);
    const float var = s2 * (1.f / H_DIM) - mu * mu;
    const float rs = rsqrtf(var + 1e-5f);
    for (int j = 0; j < H_DIM; ++j) {
      const float v = hbuf[tid][j] + b2[j];
      const float y = (v - mu) * rs * g2[j] + be2[j];
      store_A_elem((_Float16*)x2f, tid, j, (_Float16)(y / (1.f + expf(-y))));
    }
  }
  __syncthreads();

  // -- 7) GEMM3: (16x64) x (64x224) + b3, 14 N-tiles over 4 waves -----------
  for (int t = wave; t < D_OUT3 / 16; t += 4)
    wmma_tile<2>(x2f, W3f, t, &mlp_s[0][0], D_OUT3, lane, b3);

  // -- 8) drain the async wigner stream, then one barrier -------------------
#if USE_ASYNC_LDS
  __builtin_amdgcn_s_wait_asynccnt(0);
#endif
  __syncthreads();   // mlp_s + wall visible

  // -- 9) einsum (49x7 . 7x32) + scatter-add for all 16 edges ---------------
  const float inv_rescale = 1.f / RESCALE_F;
  for (int idx = tid; idx < EDGES_PER_BLOCK * OUT_PER_NODE;
       idx += BLOCK_THREADS) {
    const int el = idx / OUT_PER_NODE;
    const int o  = idx - el * OUT_PER_NODE;
    const int node = tnode[el];
    if (node >= 0) {
      const int i = o >> 5, c = o & 31;
      float acc = 0.f;
#pragma unroll
      for (int l = 0; l < M0_ROWS; ++l)
        acc += wall[el][i * 8 + l] * mlp_s[el][l * C_SPH + c];
      atomicAdd(out + (size_t)node * OUT_PER_NODE + o, acc * inv_rescale);
    }
  }
}

// ---------------------------------------------------------------------------
extern "C" void kernel_launch(void* const* d_in, const int* in_sizes, int n_in,
                              void* d_out, int out_size, void* d_ws, size_t ws_size,
                              hipStream_t stream) {
  const int*   atomic_numbers = (const int*)  d_in[0];
  const int*   edge_index     = (const int*)  d_in[1];
  const float* edge_distance  = (const float*)d_in[2];
  const float* wigner_inv     = (const float*)d_in[3];
  const float* src_emb        = (const float*)d_in[4];
  const float* tgt_emb        = (const float*)d_in[5];
  const float* W1  = (const float*)d_in[6];
  const float* b1  = (const float*)d_in[7];
  const float* g1  = (const float*)d_in[8];
  const float* be1 = (const float*)d_in[9];
  const float* W2  = (const float*)d_in[10];
  const float* b2  = (const float*)d_in[11];
  const float* g2  = (const float*)d_in[12];
  const float* be2 = (const float*)d_in[13];
  const float* W3  = (const float*)d_in[14];
  const float* b3  = (const float*)d_in[15];

  float* out = (float*)d_out;
  _Float16* wsh = (_Float16*)d_ws;   // ~70 KB scratch for f16 weight fragments

  const int E = in_sizes[1] / 2;     // edge_index is (2, E)

  // output accumulates via atomics -> zero it (async memset is capturable)
  (void)hipMemsetAsync(d_out, 0, (size_t)out_size * sizeof(float), stream);

  // stage f16 weight fragments
  {
    const int nt = 256;
    const int nb = (WS_HALVES + nt - 1) / nt;
    prep_weights_f16<<<nb, nt, 0, stream>>>(W1, W2, W3, wsh);
  }

  // fused pipeline: one block per 16 edges
  const int nb = (E + EDGES_PER_BLOCK - 1) / EDGES_PER_BLOCK;
  edge_degree_fused<<<nb, BLOCK_THREADS, 0, stream>>>(
      atomic_numbers, edge_index, edge_distance, wigner_inv, src_emb, tgt_emb,
      b1, g1, be1, b2, g2, be2, b3, wsh, out, E);
}
